// BlockR_86045374808442
// MI455X (gfx1250) — matrix operand order repaired and checked by reading, not compiled, not run on hardware
//
#include <hip/hip_runtime.h>
#include <hip/hip_bf16.h>

// ---------------------------------------------------------------------------
// CDNA5 WMMA (wave32) types: bf16 A/B fragments (16 bf16 = 8 VGPRs),
// f32 C/D accumulators (8 VGPRs).
// ---------------------------------------------------------------------------
typedef __attribute__((ext_vector_type(16))) __bf16 v16bf;
typedef __attribute__((ext_vector_type(8)))  float  v8f;

// GCC-style int4 vector: matches the async-to-LDS builtin's pointee type
// ('__attribute__((__vector_size__(4*sizeof(int)))) int') per hipcc diagnostic.
typedef int v4i __attribute__((vector_size(16)));
typedef __attribute__((address_space(1))) v4i v4i_glb;
typedef __attribute__((address_space(3))) v4i v4i_lds;

union FragBF {
    v16bf    v;
    unsigned u[8];
};

#define EMB 1024
#define HID 4096
#define BATCH 2
#define SEQ 4096
#define MROWS (BATCH * SEQ)   // 8192
#define EPS 1e-6f

// Async global->LDS copy path (gfx1250): guarded so compilation never breaks.
#ifdef __has_builtin
#if __has_builtin(__builtin_amdgcn_global_load_async_to_lds_b128) && \
    __has_builtin(__builtin_amdgcn_s_wait_asynccnt)
#define USE_ASYNC_LDS 1
#endif
#endif

// round-to-nearest-even fp32 -> bf16 (raw bits)
__device__ __forceinline__ unsigned short f2bf(float f) {
    unsigned u = __float_as_uint(f);
    u += 0x7FFFu + ((u >> 16) & 1u);
    return (unsigned short)(u >> 16);
}

__device__ __forceinline__ unsigned ld32(const unsigned short* p) {
    return *(const unsigned*)p;
}

// ---------------------------------------------------------------------------
// K0: fp32 -> bf16 cast (weights). float4 -> uint2.
// ---------------------------------------------------------------------------
__global__ void cvt_bf16_kernel(const float* __restrict__ in,
                                unsigned short* __restrict__ out, int n) {
    int i = (blockIdx.x * blockDim.x + threadIdx.x) * 4;
    if (i + 3 >= n) return;
    float4 v = *(const float4*)(in + i);
    uint2 o;
    o.x = (unsigned)f2bf(v.x) | ((unsigned)f2bf(v.y) << 16);
    o.y = (unsigned)f2bf(v.z) | ((unsigned)f2bf(v.w) << 16);
    *(uint2*)(out + i) = o;
}

// ---------------------------------------------------------------------------
// K1/K4: per-row RMS norm, fp32 in -> bf16 out. One block (8 waves) per row.
// ---------------------------------------------------------------------------
__global__ void rmsnorm_rows_kernel(const float* __restrict__ in,
                                    unsigned short* __restrict__ out,
                                    int ncols) {
    const int row = blockIdx.x;
    const float* rp = in + (size_t)row * ncols;

    float ss = 0.f;
    for (int c = threadIdx.x * 4; c < ncols; c += blockDim.x * 4) {
        float4 v = *(const float4*)(rp + c);
        ss += v.x * v.x + v.y * v.y + v.z * v.z + v.w * v.w;
    }
    #pragma unroll
    for (int off = 16; off > 0; off >>= 1)
        ss += __shfl_down(ss, off, 32);

    __shared__ float part[8];
    const int lane = threadIdx.x & 31;
    const int wv   = threadIdx.x >> 5;
    if (lane == 0) part[wv] = ss;
    __syncthreads();
    if (threadIdx.x == 0) {
        float t = 0.f;
        #pragma unroll
        for (int i = 0; i < 8; ++i) t += part[i];
        part[0] = rsqrtf(t / (float)ncols + EPS);
    }
    __syncthreads();
    const float scale = part[0];

    unsigned short* op = out + (size_t)row * ncols;
    for (int c = threadIdx.x * 4; c < ncols; c += blockDim.x * 4) {
        float4 v = *(const float4*)(rp + c);
        uint2 o;
        o.x = (unsigned)f2bf(v.x * scale) | ((unsigned)f2bf(v.y * scale) << 16);
        o.y = (unsigned)f2bf(v.z * scale) | ((unsigned)f2bf(v.w * scale) << 16);
        *(uint2*)(op + c) = o;
    }
}

// ---------------------------------------------------------------------------
// K2/K5: tiled bf16 WMMA GEMM:  C[m,n] = sum_k A[m,k] * W[n,k]  (+ residual)
// A: [M,K] bf16 row-major, W: [N,K] bf16 row-major, C: [M,N] fp32.
// Compile-time N_, K_ so all strides strength-reduce to shifts.
//
// Block: 256 threads (8 wave32s). Block tile 128(M) x 64(N), K step 32.
// Waves arranged 4(M) x 2(N); each wave computes a 32x32 tile:
//   4 v_wmma_f32_16x16x32_bf16 per K-step from 2 A-frags + 2 B-frags.
// Staging: one b128 per thread per tile slice, async-to-LDS (ASYNCcnt) when
// the gfx1250 builtin is available, else plain b128 copy. LDS rows padded to
// 40 elements (80 B): 16B-aligned chunk stores, 20-bank row stride ->
// conflict-free 16-lane fragment reads.
// ---------------------------------------------------------------------------
#define LSTR 40

template <int N_, int K_>
__global__ __launch_bounds__(256)
void gemm_bf16_wmma_kernel(const unsigned short* __restrict__ A,
                           const unsigned short* __restrict__ W,
                           const float* __restrict__ resid,
                           float* __restrict__ C) {
    __shared__ unsigned short As[128 * LSTR];
    __shared__ unsigned short Ws[64 * LSTR];

    const int n0  = blockIdx.x * 64;
    const int m0  = blockIdx.y * 128;
    const int tid = threadIdx.x;

    const int lane = tid & 31;
    const int wave = tid >> 5;
    const int wm   = wave >> 1;          // 0..3 : M sub-block of 32
    const int wn   = wave & 1;           // 0..1 : N sub-block of 32

    v8f c00 = {}, c01 = {}, c10 = {}, c11 = {};

    // --- per-thread staging addresses (constant across K loop, += 32 elems) ---
    const int srow = tid >> 2;           // 0..63
    const int scol = (tid & 3) << 3;     // 0,8,16,24
    const unsigned short* pa0 = A + (size_t)(m0 + srow) * K_ + scol;
    const unsigned short* pa1 = pa0 + (size_t)64 * K_;
    const unsigned short* pw  = W + (size_t)(n0 + srow) * K_ + scol;
    unsigned short* la0 = &As[srow * LSTR + scol];
    unsigned short* la1 = la0 + 64 * LSTR;
    unsigned short* lw  = &Ws[srow * LSTR + scol];

    const int nlocal = lane & 15;
    const int khalfB = (lane < 16) ? 0 : 16;   // B frag K base
    const int khalfA = (lane < 16) ? 0 : 8;    // unused; A frags use khalfB

    for (int kb = 0; kb < K_; kb += 32) {
        // ---- stage 128x32 A tile + 64x32 W tile (one b128 per slice) ----
#ifdef USE_ASYNC_LDS
        __builtin_amdgcn_global_load_async_to_lds_b128(
            (v4i_glb*)pa0, (v4i_lds*)la0, 0, 0);
        __builtin_amdgcn_global_load_async_to_lds_b128(
            (v4i_glb*)pa1, (v4i_lds*)la1, 0, 0);
        __builtin_amdgcn_global_load_async_to_lds_b128(
            (v4i_glb*)pw, (v4i_lds*)lw, 0, 0);
        __builtin_amdgcn_s_wait_asynccnt(0);
#else
        *(uint4*)la0 = *(const uint4*)pa0;
        *(uint4*)la1 = *(const uint4*)pa1;
        *(uint4*)lw  = *(const uint4*)pw;
#endif
        pa0 += 32; pa1 += 32; pw += 32;

        // prefetch next K tile into cache while we compute
        if (kb + 32 < K_) {
            __builtin_prefetch(pa0, 0, 1);
            __builtin_prefetch(pa1, 0, 1);
            __builtin_prefetch(pw, 0, 1);
        }
        __syncthreads();

        // ---- fragments: A and B both in contiguous K order (k = 16*(lane/16) + s) ----
        FragBF fa0, fa1, fb0, fb1;
        #pragma unroll
        for (int i = 0; i < 8; ++i) {
            // B (32x16, element (k,n) = W[n][k]): lanes 0-15 K=0..15, 16-31 K=16..31
            fb0.u[i] = ld32(&Ws[(wn * 32 +      nlocal) * LSTR + khalfB + 2 * i]);
            fb1.u[i] = ld32(&Ws[(wn * 32 + 16 + nlocal) * LSTR + khalfB + 2 * i]);
            // A (16x32): lanes 0-15 K=0..15 ; lanes 16-31 K=16..31
            const int k0 = khalfB + 2 * i;
            fa0.u[i] = ld32(&As[(wm * 32 +      nlocal) * LSTR + k0]);
            fa1.u[i] = ld32(&As[(wm * 32 + 16 + nlocal) * LSTR + k0]);
        }

        c00 = __builtin_amdgcn_wmma_f32_16x16x32_bf16(false, fa0.v, false, fb0.v,
                                                      (short)0, c00, false, false);
        c01 = __builtin_amdgcn_wmma_f32_16x16x32_bf16(false, fa0.v, false, fb1.v,
                                                      (short)0, c01, false, false);
        c10 = __builtin_amdgcn_wmma_f32_16x16x32_bf16(false, fa1.v, false, fb0.v,
                                                      (short)0, c10, false, false);
        c11 = __builtin_amdgcn_wmma_f32_16x16x32_bf16(false, fa1.v, false, fb1.v,
                                                      (short)0, c11, false, false);
        __syncthreads();
    }

    // ---- store D (VGPR v: lanes 0-15 -> M=v, lanes 16-31 -> M=v+8) ----
    const int mhalf = (lane < 16) ? 0 : 8;
    const int ngA   = n0 + wn * 32 + nlocal;
    const int ngB   = ngA + 16;
    #pragma unroll
    for (int v = 0; v < 8; ++v) {
        const int mg0 = m0 + wm * 32 + v + mhalf;
        const int mg1 = mg0 + 16;
        size_t i00 = (size_t)mg0 * N_ + ngA;
        size_t i01 = (size_t)mg0 * N_ + ngB;
        size_t i10 = (size_t)mg1 * N_ + ngA;
        size_t i11 = (size_t)mg1 * N_ + ngB;
        float o00 = c00[v], o01 = c01[v], o10 = c10[v], o11 = c11[v];
        if (resid) {
            o00 += resid[i00]; o01 += resid[i01];
            o10 += resid[i10]; o11 += resid[i11];
        }
        C[i00] = o00; C[i01] = o01; C[i10] = o10; C[i11] = o11;
    }
}

// ---------------------------------------------------------------------------
// K3: causal quadratic bar, stable form. One thread per (batch, channel h):
// running logsumexp of a and b along T, y = exp(la + lb - 2 log t), written
// in place over the `a` buffer. Next-step loads issued before the dependent
// update (software pipelining).
// ---------------------------------------------------------------------------
__global__ void scan_bar_kernel(const float* __restrict__ b_in,
                                float* __restrict__ a_io /* a in, y out */) {
    const int gid = blockIdx.x * blockDim.x + threadIdx.x;  // 0 .. BATCH*HID-1
    if (gid >= BATCH * HID) return;
    const int batch = gid / HID;
    const int h     = gid % HID;

    float*       pa = a_io + (size_t)batch * SEQ * HID + h;
    const float* pb = b_in + (size_t)batch * SEQ * HID + h;

    float ma = -3.0e38f, sa = 0.f;
    float mb = -3.0e38f, sb = 0.f;

    float va = pa[0];
    float vb = pb[0];
    for (int t = 0; t < SEQ; ++t) {
        float va_n = 0.f, vb_n = 0.f;
        if (t + 1 < SEQ) {
            va_n = pa[(size_t)(t + 1) * HID];
            vb_n = pb[(size_t)(t + 1) * HID];
        }
        if (va > ma) { sa = sa * __expf(ma - va) + 1.f; ma = va; }
        else         { sa += __expf(va - ma); }
        if (vb > mb) { sb = sb * __expf(mb - vb) + 1.f; mb = vb; }
        else         { sb += __expf(vb - mb); }

        const float la = ma + __logf(sa);
        const float lb = mb + __logf(sb);
        const float lt = __logf((float)(t + 1));
        pa[(size_t)t * HID] = __expf(la + lb - 2.f * lt);

        va = va_n;
        vb = vb_n;
    }
}

// ---------------------------------------------------------------------------
// Launch. Workspace layout (bytes):
//   r_bf16 : 16 MiB | W1b,W2b,W3b : 8 MiB each | a : 128 MiB | b : 128 MiB
// y (f32) overwrites a; y_norm (bf16) reuses b.
// ---------------------------------------------------------------------------
extern "C" void kernel_launch(void* const* d_in, const int* in_sizes, int n_in,
                              void* d_out, int out_size, void* d_ws, size_t ws_size,
                              hipStream_t stream) {
    const float* x  = (const float*)d_in[0];
    const float* W1 = (const float*)d_in[1];
    const float* W2 = (const float*)d_in[2];
    const float* W3 = (const float*)d_in[3];
    float* out = (float*)d_out;

    char* ws = (char*)d_ws;
    unsigned short* r_bf = (unsigned short*)(ws);
    unsigned short* w1b  = (unsigned short*)(ws + (size_t)16777216);
    unsigned short* w2b  = (unsigned short*)(ws + (size_t)25165824);
    unsigned short* w3b  = (unsigned short*)(ws + (size_t)33554432);
    float*          a_f  = (float*)(ws + (size_t)41943040);
    float*          b_f  = (float*)(ws + (size_t)176160768);
    unsigned short* ynb  = (unsigned short*)b_f;   // reuse after scan

    const int NW = HID * EMB;   // 4,194,304 elements per weight matrix

    // K0: cast weights to bf16
    cvt_bf16_kernel<<<NW / 1024, 256, 0, stream>>>(W1, w1b, NW);
    cvt_bf16_kernel<<<NW / 1024, 256, 0, stream>>>(W2, w2b, NW);
    cvt_bf16_kernel<<<NW / 1024, 256, 0, stream>>>(W3, w3b, NW);

    // K1: r = rmsnorm(x) -> bf16
    rmsnorm_rows_kernel<<<MROWS, 256, 0, stream>>>(x, r_bf, EMB);

    // K2: a = r @ W1^T ; b = r @ W2^T  (bf16 WMMA, f32 accum)
    {
        dim3 grid(HID / 64, MROWS / 128);
        gemm_bf16_wmma_kernel<HID, EMB><<<grid, 256, 0, stream>>>(r_bf, w1b, nullptr, a_f);
        gemm_bf16_wmma_kernel<HID, EMB><<<grid, 256, 0, stream>>>(r_bf, w2b, nullptr, b_f);
    }

    // K3: causal scan -> y (overwrites a_f)
    scan_bar_kernel<<<(BATCH * HID) / 256, 256, 0, stream>>>(b_f, a_f);

    // K4: rmsnorm(y) -> bf16 (reuses b buffer)
    rmsnorm_rows_kernel<<<MROWS, 256, 0, stream>>>(a_f, ynb, HID);

    // K5: out = x + y_norm @ W3^T
    {
        dim3 grid(EMB / 64, MROWS / 128);
        gemm_bf16_wmma_kernel<EMB, HID><<<grid, 256, 0, stream>>>(ynb, w3b, x, out);
    }
}
